// RNNModule_75823352643883
// MI455X (gfx1250) — matrix operand order repaired and checked by reading, not compiled, not run on hardware
//
#include <hip/hip_runtime.h>
#include <stdint.h>
#include <stddef.h>

#define BATCH 64
#define SEQ   512
#define DIN   512
#define HID   1024
#define NOUT  2
#define NGB   32            // GEMM blocks: block b -> mt = b&3, 8 waves cover 8 N-tiles
#define NBLK  (NGB + 1)     // +1 dedicated y-projection block
#define BH    (BATCH * HID)

typedef __attribute__((ext_vector_type(16))) __bf16   v16bf;
typedef __attribute__((ext_vector_type(8)))  float    v8f;
typedef __attribute__((ext_vector_type(4)))  unsigned uv4;
typedef __attribute__((ext_vector_type(4)))  float    fv4;

union Frag { uv4 u[2]; unsigned short us[16]; v16bf v; };
union Acc  { v8f v; float f[8]; fv4 q[2]; };

__device__ __forceinline__ unsigned short f2bf(float f) {
    union { __bf16 b; unsigned short u; } v;
    v.b = (__bf16)f;                      // native RTNE convert op
    return v.u;
}
__device__ __forceinline__ float bf2f(unsigned short s) {
    union { unsigned u; float f; } v; v.u = ((unsigned)s) << 16;
    return v.f;
}
__device__ __forceinline__ float fast_sig(float x)  { return 1.0f / (1.0f + __expf(-x)); }
__device__ __forceinline__ float fast_tanh(float x) {
    float e = __expf(2.0f * x);
    return (e - 1.0f) / (e + 1.0f);
}

// ---- CDNA5 async global->LDS copy (ASYNCcnt path), per CDNA5 ISA ----
__device__ __forceinline__ void async_ld16(unsigned lds_off, const void* gptr) {
    asm volatile("global_load_async_to_lds_b128 %0, %1, off"
                 :: "v"(lds_off), "v"((unsigned long long)(size_t)gptr)
                 : "memory");
}
__device__ __forceinline__ void wait_async0() {
    asm volatile("s_wait_asynccnt 0x0" ::: "memory");
}

// ---------------- grid-wide barrier (arrive + generation flip) ----------------
__device__ __forceinline__ void gsync(unsigned* bar, unsigned nb) {
    __syncthreads();
    if (threadIdx.x == 0) {
        unsigned gen = __hip_atomic_load(bar + 1, __ATOMIC_RELAXED, __HIP_MEMORY_SCOPE_AGENT);
        __threadfence();
        unsigned arr = __hip_atomic_fetch_add(bar, 1u, __ATOMIC_ACQ_REL, __HIP_MEMORY_SCOPE_AGENT);
        if (arr == nb - 1u) {
            __hip_atomic_store(bar, 0u, __ATOMIC_RELAXED, __HIP_MEMORY_SCOPE_AGENT);
            __hip_atomic_fetch_add(bar + 1, 1u, __ATOMIC_RELEASE, __HIP_MEMORY_SCOPE_AGENT);
        } else {
            while (__hip_atomic_load(bar + 1, __ATOMIC_ACQUIRE, __HIP_MEMORY_SCOPE_AGENT) == gen)
                __builtin_amdgcn_s_sleep(1);
        }
    }
    __syncthreads();
}

// ---------------- f32 -> bf16 weight conversion ----------------
__global__ void cvt_f32_bf16(const float* __restrict__ src, unsigned short* __restrict__ dst, int n) {
    int i = blockIdx.x * blockDim.x + threadIdx.x;
    if (i < n) dst[i] = f2bf(src[i]);
}

__global__ void init_barrier(unsigned* bar) {
    if (threadIdx.x < 2) bar[threadIdx.x] = 0u;
}

// ---------------- xproj: [L*B,512] x [512,1024] via bf16 WMMA ----------------
// Each wave: 16 rows x 64 cols (4 N-tiles) -> one A conversion feeds 4 WMMAs.
// Output stored transposed: xp[l][n][b]  (scan reads contiguous b128 per lane)
__global__ __launch_bounds__(256) void xproj_kernel(
    const float* __restrict__ x, const unsigned short* __restrict__ WxBf,
    const float* __restrict__ Wxb, float* __restrict__ xp)
{
    const int w    = blockIdx.x * 8 + (threadIdx.x >> 5);   // 32768 waves
    const int lane = threadIdx.x & 31;
    const int ng = w & 15;          // 16 N-groups of 64 cols
    const int mt = w >> 4;          // 2048 M-tiles
    const int l  = mt >> 2;         // 4 M-tiles per timestep (B=64)
    const int b0 = (mt & 3) << 4;
    const int ln = lane & 15;
    const int hi = lane >> 4;

    const float* xrow = x + ((size_t)(b0 + ln) * SEQ + l) * DIN;   // x[b][l][:]
    const unsigned short* wrow[4];
#pragma unroll
    for (int j = 0; j < 4; ++j)
        wrow[j] = WxBf + (size_t)(ng * 64 + j * 16 + ln) * DIN;    // Wx[n][:]

    Acc acc[4];
#pragma unroll
    for (int j = 0; j < 4; ++j)
#pragma unroll
        for (int i = 0; i < 8; ++i) acc[j].f[i] = 0.0f;

    for (int k0 = 0; k0 < DIN; k0 += 32) {
        Frag A;
        const float* pa0 = xrow + k0 + hi * 8;        // K groups per ISA A-layout
        const float* pa1 = xrow + k0 + 16 + hi * 8;
#pragma unroll
        for (int i = 0; i < 8; ++i) { A.us[i] = f2bf(pa0[i]); A.us[8 + i] = f2bf(pa1[i]); }
#pragma unroll
        for (int j = 0; j < 4; ++j) {
            Frag Bm;
            const uv4* pb = (const uv4*)(wrow[j] + k0 + hi * 8);
            Bm.u[0] = pb[0]; Bm.u[1] = pb[2];
            acc[j].v = __builtin_amdgcn_wmma_f32_16x16x32_bf16(
                false, A.v, false, Bm.v, (short)0, acc[j].v, false, false);
        }
    }

#pragma unroll
    for (int j = 0; j < 4; ++j) {
        const int col  = ng * 64 + j * 16 + ln;
        const float bn = Wxb[col];
        float* dst = xp + ((size_t)l * HID + col) * BATCH + b0 + hi * 8;
        fv4 s0, s1;
        s0.x = acc[j].f[0] + bn; s0.y = acc[j].f[1] + bn;
        s0.z = acc[j].f[2] + bn; s0.w = acc[j].f[3] + bn;
        s1.x = acc[j].f[4] + bn; s1.y = acc[j].f[5] + bn;
        s1.z = acc[j].f[6] + bn; s1.w = acc[j].f[7] + bn;
        ((fv4*)dst)[0] = s0; ((fv4*)dst)[1] = s1;
    }
}

// ---------------- per-step tiny output projection (runs one step behind) ----------------
__device__ __forceinline__ void y_step(const unsigned short* __restrict__ h,
                                       const unsigned short* __restrict__ WaBf,
                                       const float* __restrict__ Wab,
                                       float* __restrict__ out, int t)
{
    const int tid = threadIdx.x;
    if (tid >= BATCH * NOUT) return;
    const int b = tid >> 1, o = tid & 1;
    const unsigned short* hr = h    + (size_t)b * HID;
    const unsigned short* wr = WaBf + (size_t)o * HID;
    float acc = 0.0f;
    for (int k = 0; k < HID; k += 8) {
        uv4 hv = *(const uv4*)(hr + k);
        uv4 wv = *(const uv4*)(wr + k);
        const unsigned short* hs = (const unsigned short*)&hv;
        const unsigned short* ws = (const unsigned short*)&wv;
#pragma unroll
        for (int i = 0; i < 8; ++i) acc += bf2f(hs[i]) * bf2f(ws[i]);
    }
    out[((size_t)b * SEQ + t) * NOUT + o] = fast_sig(acc + Wab[o]);
}

// ---------------- persistent recurrent scan ----------------
// Block b (b<32): mt = b&3 (16 batch rows), 8 waves cover N-tiles (b>>2)*8 + wave.
// Per step: async-stage the block's 16x1024 bf16 A-panel into LDS, GEMM from LDS,
// cell state c entirely in registers, h double-buffered bf16 in global (L2-resident).
__global__ __launch_bounds__(256) void lstm_scan_kernel(
    const unsigned short* __restrict__ WhBf, const float* __restrict__ Whb,
    const float* __restrict__ xproj, const unsigned short* __restrict__ WaBf,
    const float* __restrict__ Wab, unsigned short* hbuf,
    unsigned* bar, float* __restrict__ out)
{
    __shared__ unsigned short hstage[16 * HID];   // 32 KB A-panel

    const bool isY  = (blockIdx.x == NGB);
    const int  lane = threadIdx.x & 31;
    const int  ln = lane & 15, hi = lane >> 4;

    int mt = 0, col = 0, drow0 = 0;
    float biasn = 0.0f;
    Acc creg;
#pragma unroll
    for (int i = 0; i < 8; ++i) creg.f[i] = 0.0f;

    if (!isY) {
        mt = blockIdx.x & 3;
        const int nt = (blockIdx.x >> 2) * 8 + (threadIdx.x >> 5);  // 0..63
        col   = nt * 16 + ln;
        drow0 = mt * 16 + hi * 8;
        biasn = Whb[col];
        // zero h_0 (buffer 0): tiles partition (64 x 1024) -> full coverage
#pragma unroll
        for (int r = 0; r < 8; ++r) hbuf[(size_t)(drow0 + r) * HID + col] = 0;
    }
    gsync(bar, NBLK);

    const unsigned lds_base = (unsigned)(size_t)&hstage[0];   // low 32 bits == LDS offset

    for (int t = 0; t < SEQ; ++t) {
        if (!isY) {
            const unsigned short* hsrc = hbuf + (size_t)(t & 1) * BH;
            unsigned short*       hdst = hbuf + (size_t)((t + 1) & 1) * BH;

            // ---- async stage A-panel: rows [mt*16, mt*16+16) x HID bf16 = 32KB ----
            const char* psrc = (const char*)(hsrc + (size_t)(mt * 16) * HID);
#pragma unroll
            for (int j = 0; j < 8; ++j) {
                const int c = threadIdx.x * 8 + j;          // 2048 x 16B chunks
                async_ld16(lds_base + c * 16, psrc + (size_t)c * 16);
            }
            wait_async0();
            __syncthreads();

            const unsigned short* bptr = WhBf + (size_t)col * HID;
            Acc acc;
#pragma unroll
            for (int i = 0; i < 8; ++i) acc.f[i] = 0.0f;

            for (int k0 = 0; k0 < HID; k0 += 32) {
                Frag A, Bm;
                A.u[0] = *(const uv4*)(&hstage[ln * HID + k0 + hi * 8]);
                A.u[1] = *(const uv4*)(&hstage[ln * HID + k0 + 16 + hi * 8]);
                const uv4* pb = (const uv4*)(bptr + k0 + hi * 8);
                Bm.u[0] = pb[0]; Bm.u[1] = pb[2];
                if (k0 + 256 < HID) __builtin_prefetch(bptr + k0 + 256, 0, 1);
                acc.v = __builtin_amdgcn_wmma_f32_16x16x32_bf16(
                    false, A.v, false, Bm.v, (short)0, acc.v, false, false);
            }

            // xproj tile: contiguous 8 floats per lane ([L][H][B] layout)
            const float* xq = xproj + ((size_t)t * HID + col) * BATCH + drow0;
            fv4 x0 = ((const fv4*)xq)[0], x1 = ((const fv4*)xq)[1];
            float xv[8] = { x0.x, x0.y, x0.z, x0.w, x1.x, x1.y, x1.z, x1.w };

#pragma unroll
            for (int r = 0; r < 8; ++r) {
                float a  = acc.f[r] + biasn + xv[r];
                float s  = fast_sig(a);
                float g  = fast_tanh(a);
                float cv = s * (creg.f[r] + g);     // s*c + s*g
                creg.f[r] = cv;
                float hv = s * fast_tanh(cv);
                hdst[(size_t)(drow0 + r) * HID + col] = f2bf(hv);
            }
        } else if (t >= 1) {
            // y_{t-1} from h_t (buffer t&1), overlapped with step t's GEMM
            y_step(hbuf + (size_t)(t & 1) * BH, WaBf, Wab, out, t - 1);
        }
        gsync(bar, NBLK);
    }
    if (isY) y_step(hbuf + (size_t)(SEQ & 1) * BH, WaBf, Wab, out, SEQ - 1);
}

// ---------------- host ----------------
extern "C" void kernel_launch(void* const* d_in, const int* in_sizes, int n_in,
                              void* d_out, int out_size, void* d_ws, size_t ws_size,
                              hipStream_t stream) {
    (void)in_sizes; (void)n_in; (void)out_size; (void)ws_size;
    const float* x    = (const float*)d_in[0];
    const float* Whw  = (const float*)d_in[1];
    const float* Whb  = (const float*)d_in[2];
    const float* Wxw  = (const float*)d_in[3];
    const float* Wxb  = (const float*)d_in[4];
    const float* Waw  = (const float*)d_in[5];
    const float* Wab  = (const float*)d_in[6];
    float* out = (float*)d_out;

    char* ws = (char*)d_ws;
    unsigned short* WhBf = (unsigned short*)(ws);                              // 2 MB
    unsigned short* WxBf = (unsigned short*)(ws + (2ull << 20));               // 1 MB
    unsigned short* WaBf = (unsigned short*)(ws + (3ull << 20));               // 4 KB
    unsigned short* hbuf = (unsigned short*)(ws + (3ull << 20) + 65536);       // 256 KB (2 bufs)
    unsigned*       bar  = (unsigned*)      (ws + (3ull << 20) + 65536 + 2ull * BH * sizeof(unsigned short));
    float*          xprj = (float*)         (ws + (4ull << 20));               // 128 MB

    cvt_f32_bf16<<<(HID * HID + 255) / 256, 256, 0, stream>>>(Whw, WhBf, HID * HID);
    cvt_f32_bf16<<<(HID * DIN + 255) / 256, 256, 0, stream>>>(Wxw, WxBf, HID * DIN);
    cvt_f32_bf16<<<(NOUT * HID + 255) / 256, 256, 0, stream>>>(Waw, WaBf, NOUT * HID);
    init_barrier<<<1, 32, 0, stream>>>(bar);

    // 2048 M-tiles x 16 N-groups = 32768 waves / 8 waves per block
    xproj_kernel<<<4096, 256, 0, stream>>>(x, WxBf, Wxb, xprj);

    lstm_scan_kernel<<<NBLK, 256, 0, stream>>>(WhBf, Whb, xprj, WaBf, Wab, hbuf, bar, out);
}
